// GCN_Layer_5248450036422
// MI455X (gfx1250) — compile-verified, matrix operand order checked by reading, not run on hardware
//
#include <hip/hip_runtime.h>
#include <hip/hip_bf16.h>

// ---------------------------------------------------------------------------
// GCN layer: COO SpMM (atomic scatter) -> in-place [*,64]@[64,64] GEMM (bf16
// split-precision WMMA, fp32 accumulate) + sigmoid, fused mean over behaviors.
// Split:  x = hi + lo (both bf16, truncation; lo = x - hi is exact in f32)
//         Y ~= Ahi*Bhi + Ahi*Blo + Alo*Bhi     (error ~2^-16 relative)
// ---------------------------------------------------------------------------

#define NUv 100000
#define NIv 200000
#define Bv  3
#define Ev  1000000
#define Dv  64

typedef __attribute__((ext_vector_type(16))) __bf16         v16bf;
typedef __attribute__((ext_vector_type(16))) unsigned short v16us;
typedef __attribute__((ext_vector_type(8)))  unsigned short v8us;
typedef __attribute__((ext_vector_type(8)))  float          v8f;

#define LSTRIDE 72   // ushort row stride: 144B = 16B-aligned, conflict-free

// sigmoid with hardware v_exp_f32 + v_rcp_f32 (no IEEE divide sequence)
__device__ __forceinline__ float sigmoidf_fast(float x) {
    return __builtin_amdgcn_rcpf(1.0f + __expf(-x));
}

// ---------------------------------------------------------------------------
__global__ void zero_kernel(float4* __restrict__ p, long long n4) {
    long long i = (long long)blockIdx.x * blockDim.x + threadIdx.x;
    if (i < n4) p[i] = make_float4(0.f, 0.f, 0.f, 0.f);
}

// ---------------------------------------------------------------------------
// COO SpMM scatter: acc[b, rows[e], :] += vals[e] * dense[cols[e], :]
// One wave per edge; lane handles 2 consecutive floats (256B coalesced gather
// per wave, 2x global_atomic_add_f32 per lane).
// ---------------------------------------------------------------------------
__global__ void spmm_atomic_kernel(const float* __restrict__ vals,
                                   const int*   __restrict__ rows,
                                   const int*   __restrict__ cols,
                                   const float* __restrict__ dense,
                                   float*       __restrict__ acc,
                                   unsigned int total_threads,
                                   int nrows_out) {
    unsigned int t = blockIdx.x * 256u + threadIdx.x;
    if (t >= total_threads) return;
    unsigned int e = t >> 5;                 // edge id in [0, Bv*Ev)
    int p = (int)(t & 31u);
    unsigned int b = e / (unsigned int)Ev;   // behavior

    float v = vals[e];
    int r = rows[e];
    int c = cols[e];

    const float* dp = dense + (long long)c * Dv + p * 2;
    float2 d = *(const float2*)dp;

    float* ap = acc + ((long long)b * nrows_out + r) * Dv + p * 2;
    unsafeAtomicAdd(ap,     v * d.x);
    unsafeAtomicAdd(ap + 1, v * d.y);
}

// ---------------------------------------------------------------------------
// Load a 16-element bf16 operand fragment: two aligned 16B runs from LDS.
// Lane layout (A and B identical): lanes 0-15 -> K {0..7,16..23},
// lanes 16-31 -> K {8..15,24..31} of the 32-wide k-tile.
// base = row_ptr + kt*32 + 8*half  (16B aligned by construction)
// ---------------------------------------------------------------------------
__device__ __forceinline__ v16bf load_frag(const unsigned short* base) {
    v8us a = *(const v8us*)(base);
    v8us b = *(const v8us*)(base + 16);
    v16us u = __builtin_shufflevector(a, b, 0,1,2,3,4,5,6,7,8,9,10,11,12,13,14,15);
    return __builtin_bit_cast(v16bf, u);
}

// Pack bf16(hi) of two floats into one dword via byte-perm (truncation).
__device__ __forceinline__ unsigned int pack_hi2(float x0, float x1) {
    return __builtin_amdgcn_perm(__float_as_uint(x1), __float_as_uint(x0), 0x07060302u);
}

// ---------------------------------------------------------------------------
// In-place GEMM + sigmoid, fused mean across Bv behaviors.
//   X:  [Bv, nrows, 64] fp32 accumulators -> overwritten with sigmoid(X_b @ W)
//   Mo: [nrows, 64]    <- sigmoid(mean_b(X_b @ W))
// Block = 256 threads (8 waves) over a 32-row slab: wave = (mtile<<2)|ntile.
// ---------------------------------------------------------------------------
__global__ void __launch_bounds__(256)
gemm_sigmoid_kernel(float* __restrict__ X,
                    float* __restrict__ Mo,
                    const float* __restrict__ W,
                    int nrows) {
    __shared__ __align__(16) unsigned short sWThi[Dv * LSTRIDE]; // W^T hi [n][k]
    __shared__ __align__(16) unsigned short sWTlo[Dv * LSTRIDE]; // W^T lo
    __shared__ __align__(16) unsigned short sXhi[32 * LSTRIDE];  // X slab hi
    __shared__ __align__(16) unsigned short sXlo[32 * LSTRIDE];  // X slab lo

    const int tid   = threadIdx.x;
    const int wave  = tid >> 5;
    const int lane  = tid & 31;
    const int half  = lane >> 4;
    const int idx   = lane & 15;
    const int mtile = wave >> 2;            // 0..1
    const int n0    = (wave & 3) * 16;      // 0,16,32,48
    const int r0    = blockIdx.x * 32;

    // ---- Stage W transposed as bf16 hi/lo (once per block) -----------------
    for (int i = tid; i < Dv * Dv; i += 256) {
        int k = i >> 6, n = i & 63;
        float w = W[i];
        unsigned int u = __float_as_uint(w);
        unsigned short hi = (unsigned short)(u >> 16);
        float lo = w - __uint_as_float(u & 0xFFFF0000u);
        sWThi[n * LSTRIDE + k] = hi;
        sWTlo[n * LSTRIDE + k] = (unsigned short)(__float_as_uint(lo) >> 16);
    }
    __syncthreads();

    // ---- B fragments for this wave (2 k-tiles, hi/lo) ----------------------
    const unsigned short* wrow_h = sWThi + (n0 + idx) * LSTRIDE + 8 * half;
    const unsigned short* wrow_l = sWTlo + (n0 + idx) * LSTRIDE + 8 * half;
    v16bf bhi[2], blo[2];
#pragma unroll
    for (int kt = 0; kt < 2; ++kt) {
        bhi[kt] = load_frag(wrow_h + kt * 32);
        blo[kt] = load_frag(wrow_l + kt * 32);
    }

    float csum[8];
#pragma unroll
    for (int j = 0; j < 8; ++j) csum[j] = 0.0f;

    const unsigned short* arow_h = sXhi + (mtile * 16 + idx) * LSTRIDE + 8 * half;
    const unsigned short* arow_l = sXlo + (mtile * 16 + idx) * LSTRIDE + 8 * half;

    for (int b = 0; b < Bv; ++b) {
        __syncthreads();   // previous iteration's LDS readers done
        float* Xb = X + ((long long)b * nrows + r0) * Dv;

        // load 32x64 slab from global, split hi/lo, store packed to LDS
        {
            int i0  = tid * 8;              // 256 threads x 8 = 2048 elements
            int row = i0 >> 6, col = i0 & 63;
            const float* g = Xb + row * Dv + col;
            float4 a = ((const float4*)g)[0];
            float4 bb = ((const float4*)g)[1];
            float x[8] = {a.x, a.y, a.z, a.w, bb.x, bb.y, bb.z, bb.w};
            unsigned int hp[4], lp[4];
#pragma unroll
            for (int i = 0; i < 4; ++i) {
                float x0 = x[2 * i], x1 = x[2 * i + 1];
                hp[i] = pack_hi2(x0, x1);
                float l0 = x0 - __uint_as_float(__float_as_uint(x0) & 0xFFFF0000u);
                float l1 = x1 - __uint_as_float(__float_as_uint(x1) & 0xFFFF0000u);
                lp[i] = pack_hi2(l0, l1);
            }
            unsigned int* hdst = (unsigned int*)(sXhi + row * LSTRIDE + col);
            unsigned int* ldst = (unsigned int*)(sXlo + row * LSTRIDE + col);
            ((uint4*)hdst)[0] = make_uint4(hp[0], hp[1], hp[2], hp[3]);
            ((uint4*)ldst)[0] = make_uint4(lp[0], lp[1], lp[2], lp[3]);
        }
        __syncthreads();

        v8f c = {};
#pragma unroll
        for (int kt = 0; kt < 2; ++kt) {
            v16bf ahi = load_frag(arow_h + kt * 32);
            v16bf alo = load_frag(arow_l + kt * 32);
            c = __builtin_amdgcn_wmma_f32_16x16x32_bf16(
                    false, ahi, false, bhi[kt], (short)0, c, false, false);
            c = __builtin_amdgcn_wmma_f32_16x16x32_bf16(
                    false, ahi, false, blo[kt], (short)0, c, false, false);
            c = __builtin_amdgcn_wmma_f32_16x16x32_bf16(
                    false, alo, false, bhi[kt], (short)0, c, false, false);
        }

        // C/D layout: VGPR j -> row M = j + 8*half, col N = idx
#pragma unroll
        for (int j = 0; j < 8; ++j) {
            int row = mtile * 16 + j + 8 * half;
            Xb[row * Dv + n0 + idx] = sigmoidf_fast(c[j]);
            csum[j] += c[j];
        }
    }

    float* Mb = Mo + (long long)r0 * Dv;
#pragma unroll
    for (int j = 0; j < 8; ++j) {
        int row = mtile * 16 + j + 8 * half;
        Mb[row * Dv + n0 + idx] = sigmoidf_fast(csum[j] * (1.0f / 3.0f));
    }
}

// ---------------------------------------------------------------------------
extern "C" void kernel_launch(void* const* d_in, const int* in_sizes, int n_in,
                              void* d_out, int out_size, void* d_ws, size_t ws_size,
                              hipStream_t stream) {
    const float* user_emb = (const float*)d_in[0];
    const float* item_emb = (const float*)d_in[1];
    const float* u2i_vals = (const float*)d_in[2];
    const int*   u2i_rows = (const int*)d_in[3];
    const int*   u2i_cols = (const int*)d_in[4];
    const float* i2u_vals = (const float*)d_in[5];
    const int*   i2u_rows = (const int*)d_in[6];
    const int*   i2u_cols = (const int*)d_in[7];
    const float* u_w      = (const float*)d_in[8];
    const float* i_w      = (const float*)d_in[9];

    float* out = (float*)d_out;
    float* ue  = out;                                   // [NU,64]
    float* ie  = ue  + (long long)NUv * Dv;             // [NI,64]
    float* ues = ie  + (long long)NIv * Dv;             // [B,NU,64]
    float* ies = ues + (long long)Bv * NUv * Dv;        // [B,NI,64]

    // 1) zero the SpMM accumulator regions (ues..end of ies, contiguous)
    long long zf  = (long long)Bv * (NUv + NIv) * Dv;
    long long zn4 = zf / 4;
    zero_kernel<<<(unsigned)((zn4 + 255) / 256), 256, 0, stream>>>((float4*)ues, zn4);

    // 2) SpMM scatter (one wave per edge)
    unsigned int tthreads = (unsigned int)((long long)Bv * Ev * 32);
    unsigned int sblocks  = (tthreads + 255u) / 256u;
    spmm_atomic_kernel<<<sblocks, 256, 0, stream>>>(
        u2i_vals, u2i_rows, u2i_cols, item_emb, ues, tthreads, NUv);
    spmm_atomic_kernel<<<sblocks, 256, 0, stream>>>(
        i2u_vals, i2u_rows, i2u_cols, user_emb, ies, tthreads, NIv);

    // 3) in-place GEMM + sigmoid (+ fused mean -> ue/ie)
    gemm_sigmoid_kernel<<<NUv / 32, 256, 0, stream>>>(ues, ue, u_w, NUv);
    gemm_sigmoid_kernel<<<NIv / 32, 256, 0, stream>>>(ies, ie, i_w, NIv);
}
